// MultiheadAttention_13142599926181
// MI455X (gfx1250) — matrix operand order repaired and checked by reading, not compiled, not run on hardware
//
#include <hip/hip_runtime.h>
#include <hip/hip_bf16.h>
#include <stdint.h>

// ---------------------------------------------------------------------------
// GQA multi-head attention forward for MI455X (gfx1250, wave32, WMMA + TDM).
// Pipeline: f32->bf16 convert, Wt transposes, 3 WMMA GEMM projections,
// flash-attention (online softmax) with WMMA QK^T and PV, WMMA out-proj (f32).
// All GEMM/flash tile staging uses the Tensor Data Mover (tensor_load_to_lds)
// with double-buffered LDS and TENSORcnt pipelining.
// ---------------------------------------------------------------------------

typedef __bf16 bf16_t;
typedef __bf16 v8bf  __attribute__((ext_vector_type(8)));
typedef __bf16 v16bf __attribute__((ext_vector_type(16)));
typedef float  v8f   __attribute__((ext_vector_type(8)));
typedef float  v4f   __attribute__((ext_vector_type(4)));
typedef unsigned int u32x4 __attribute__((ext_vector_type(4)));
typedef int          i32x8 __attribute__((ext_vector_type(8)));
typedef int          i32x4 __attribute__((ext_vector_type(4)));

#define EMBED 2048
#define S_LEN 2048
#define BATCH 2
#define HQ    32
#define HKV   8
#define HD    64
#define KVD   512
#define M_TOK (BATCH * S_LEN)   // 4096 tokens

static __device__ __forceinline__ bf16_t f2bf(float f) {
  union { float f; uint32_t u; } v; v.f = f;
  uint32_t r = v.u + 0x7FFFu + ((v.u >> 16) & 1u);   // round-to-nearest-even
  uint16_t h = (uint16_t)(r >> 16);
  bf16_t o; __builtin_memcpy(&o, &h, 2); return o;
}

static __device__ __forceinline__ v16bf mk16(v8bf lo, v8bf hi) {
  v16bf r;
#pragma unroll
  for (int i = 0; i < 8; ++i) { r[i] = lo[i]; r[i + 8] = hi[i]; }
  return r;
}

static __device__ __forceinline__ v8f wmma_bf16(v16bf a, v16bf b, v8f c) {
  // D = A(16x32) * B(32x16) + C(16x16 f32)
  return __builtin_amdgcn_wmma_f32_16x16x32_bf16(false, a, false, b,
                                                 (short)0, c, false, false);
}

// ---------------------------------------------------------------------------
// TDM: async 2-D bf16 tile load Global -> LDS with per-row padding.
// D# packed per CDNA5 ISA 8.3/8.4: group0 = {count, lds_addr, global_addr,
// type=2}; group1 = {data_size=2B, pad cfg, tensor dims, tile dims, stride}.
// This toolchain exposes the 6-arg builtin: (g0, g1, g2, g3, g_extra, cpol).
//   tile0   : tile width  (elements, contiguous in memory)
//   tile1   : tile height (rows)
//   stride0 : row stride of source tensor (elements)
//   padI    : pad_interval code (interval = 2<<padI DWORDs)
//   padA    : pad_amount   code (amount  = padA+1  DWORDs appended per interval)
// ---------------------------------------------------------------------------
static __device__ __forceinline__ void tdm_load_bf16_2d(
    unsigned lds_off, const bf16_t* gptr,
    unsigned tile0, unsigned tile1, unsigned stride0,
    unsigned padI, unsigned padA) {
  unsigned long long ga = (unsigned long long)(uintptr_t)gptr;
  u32x4 g0;
  g0[0] = 1u;                                  // count=1 (valid descriptor)
  g0[1] = lds_off;                             // LDS byte address
  g0[2] = (unsigned)ga;                        // global_addr[31:0]
  g0[3] = (unsigned)(ga >> 32) | (2u << 30);   // global_addr[56:32] | type=2

  const unsigned dim0 = stride0;               // >= tile0 in all our uses
  const unsigned dim1 = 1u << 20;              // generous row count (no OOB)
  i32x8 g1;
  g1[0] = (int)((1u << 16)                     // data_size = 2 bytes
              | (1u << 20)                     // pad_enable
              | (padI << 22) | (padA << 25));
  g1[1] = (int)((dim0 & 0xFFFFu) << 16);       // tensor_dim0[15:0]
  g1[2] = (int)((dim0 >> 16) | ((dim1 & 0xFFFFu) << 16));   // dim0 hi | dim1 lo
  g1[3] = (int)((dim1 >> 16) | (tile0 << 16)); // dim1 hi | tile_dim0
  g1[4] = (int)tile1;                          // tile_dim1 (tile_dim2 = 0)
  g1[5] = (int)stride0;                        // tensor_dim0_stride[31:0]
  g1[6] = 0;                                   // stride hi | dim1_stride lo
  g1[7] = 0;
  i32x4 z4 = {0, 0, 0, 0};                     // 2-D tensor: groups 2/3 unused
  i32x8 z8 = {0, 0, 0, 0, 0, 0, 0, 0};
  __builtin_amdgcn_tensor_load_to_lds(g0, g1, z4, z4, z8, 0);
}

// ---------------------------------------------------------------------------
// f32 -> bf16 elementwise convert (8 elems/thread, b128 stores)
// ---------------------------------------------------------------------------
__global__ __launch_bounds__(256)
void cvt_f32_bf16(const float* __restrict__ x, bf16_t* __restrict__ y, int n) {
  int i = (blockIdx.x * 256 + threadIdx.x) * 8;
  if (i >= n) return;
  v4f a = *(const v4f*)(x + i);
  v4f c = *(const v4f*)(x + i + 4);
  v8bf o;
#pragma unroll
  for (int j = 0; j < 4; ++j) { o[j] = f2bf(a[j]); o[j + 4] = f2bf(c[j]); }
  *(v8bf*)(y + i) = o;
}

// ---------------------------------------------------------------------------
// Transpose-convert: W[R x C] f32 row-major -> Wt[C x R] bf16 row-major
// ---------------------------------------------------------------------------
__global__ __launch_bounds__(256)
void wtrans_bf16(const float* __restrict__ W, bf16_t* __restrict__ Wt,
                 int R, int C) {
  __shared__ float tile[32][33];
  const int tx = threadIdx.x, ty = threadIdx.y;     // (32, 8)
  const int c0 = blockIdx.x * 32, r0 = blockIdx.y * 32;
#pragma unroll
  for (int j = 0; j < 4; ++j)
    tile[ty + j * 8][tx] = W[(size_t)(r0 + ty + j * 8) * C + c0 + tx];
  __syncthreads();
#pragma unroll
  for (int j = 0; j < 4; ++j)
    Wt[(size_t)(c0 + ty + j * 8) * R + r0 + tx] = f2bf(tile[tx][ty + j * 8]);
}

// ---------------------------------------------------------------------------
// WMMA GEMM: C[M x N] = A[M x 2048] (bf16 rm) * Bt[N x 2048]^T (bf16 rm)
// Block tile 128x128, BK=32, 256 thr = 8 waves (2x4), wave tile 64x32.
// A/B tiles DMA'd by TDM (wave 0) into double-buffered LDS, 40-elem stride
// (32 data + 4-DWORD pad via D# pad config) for conflict-free b128 frag reads.
// MODE epilogues: 0=Q[B,H,S,D] 1=K[B,Hkv,S,D] 2=V^T[B,Hkv,D,S] 3=f32 rm out
// ---------------------------------------------------------------------------
template <int MODE>
__global__ __launch_bounds__(256)
void gemm_bf16_wmma(const bf16_t* __restrict__ A, const bf16_t* __restrict__ Bt,
                    void* __restrict__ Out, int N) {
  constexpr int K = EMBED;
  constexpr int NT = K / 32;                       // 64 k-steps
  __shared__ alignas(16) bf16_t lA[2][128 * 40];
  __shared__ alignas(16) bf16_t lB[2][128 * 40];

  const int m0 = blockIdx.y * 128;
  const int n0 = blockIdx.x * 128;
  const int tid = threadIdx.x;
  const int lane = tid & 31;
  const int wid = tid >> 5;
  const int wm = wid >> 2;        // 0..1 -> 64 rows
  const int wn = wid & 3;         // 0..3 -> 32 cols
  const int half = lane >> 4;
  const int l16 = lane & 15;

  const bf16_t* Abase = A  + (size_t)m0 * K;
  const bf16_t* Bbase = Bt + (size_t)n0 * K;
  const unsigned ldsA0 = (unsigned)(uintptr_t)&lA[0][0];
  const unsigned ldsA1 = (unsigned)(uintptr_t)&lA[1][0];
  const unsigned ldsB0 = (unsigned)(uintptr_t)&lB[0][0];
  const unsigned ldsB1 = (unsigned)(uintptr_t)&lB[1][0];

  if (wid == 0) {   // wave 0 drives the TDM; EXEC all-ones (uniform branch)
    tdm_load_bf16_2d(ldsA0, Abase, 32, 128, K, 3, 3);   // 16-DW rows, 4-DW pad
    tdm_load_bf16_2d(ldsB0, Bbase, 32, 128, K, 3, 3);
  }

  v8f acc[4][2];
#pragma unroll
  for (int i = 0; i < 4; ++i)
#pragma unroll
    for (int j = 0; j < 2; ++j)
#pragma unroll
      for (int e = 0; e < 8; ++e) acc[i][j][e] = 0.0f;

  for (int it = 0; it < NT; ++it) {
    const int cur = it & 1;
    if (wid == 0) {
      if (it + 1 < NT) {      // prefetch next tile into free buffer, then wait
        const int kn = (it + 1) * 32;
        tdm_load_bf16_2d(cur ? ldsA0 : ldsA1, Abase + kn, 32, 128, K, 3, 3);
        tdm_load_bf16_2d(cur ? ldsB0 : ldsB1, Bbase + kn, 32, 128, K, 3, 3);
        __builtin_amdgcn_s_wait_tensorcnt(2);   // oldest 2 (current tile) done
      } else {
        __builtin_amdgcn_s_wait_tensorcnt(0);
      }
    }
    __syncthreads();          // current tile visible to all waves

    v16bf af[4], bfr[2];
#pragma unroll
    for (int mi = 0; mi < 4; ++mi) {
      // A frag 16x32: lane row = l16, K = half*8 + e(0..7), 16 + half*8 + e
      const bf16_t* r = &lA[cur][(wm * 64 + mi * 16 + l16) * 40];
      af[mi] = mk16(*(const v8bf*)(r + half * 8),
                    *(const v8bf*)(r + 16 + half * 8));
    }
#pragma unroll
    for (int ni = 0; ni < 2; ++ni) {
      // B frag 32x16: lane col = l16, K = half*8 + e(0..7), 16 + half*8 + e
      const bf16_t* r = &lB[cur][(wn * 32 + ni * 16 + l16) * 40];
      bfr[ni] = mk16(*(const v8bf*)(r + half * 8),
                     *(const v8bf*)(r + 16 + half * 8));
    }
#pragma unroll
    for (int mi = 0; mi < 4; ++mi)
#pragma unroll
      for (int ni = 0; ni < 2; ++ni)
        acc[mi][ni] = wmma_bf16(af[mi], bfr[ni], acc[mi][ni]);

    __syncthreads();          // buffer free before TDM overwrites it
  }

#pragma unroll
  for (int mi = 0; mi < 4; ++mi)
#pragma unroll
    for (int ni = 0; ni < 2; ++ni)
#pragma unroll
      for (int e = 0; e < 8; ++e) {
        const int m = m0 + wm * 64 + mi * 16 + half * 8 + e;
        const int n = n0 + wn * 32 + ni * 16 + l16;
        const float v = acc[mi][ni][e];
        if (MODE == 3) {
          ((float*)Out)[(size_t)m * N + n] = v;
        } else {
          const int b = m >> 11, s = m & (S_LEN - 1);
          const int h = n >> 6,  d = n & 63;
          bf16_t* o = (bf16_t*)Out;
          if (MODE == 0)
            o[(((size_t)b * HQ + h) * S_LEN + s) * HD + d] = f2bf(v);
          else if (MODE == 1)
            o[(((size_t)b * HKV + h) * S_LEN + s) * HD + d] = f2bf(v);
          else // MODE 2: V stored transposed [B,Hkv,D,S]
            o[(((size_t)b * HKV + h) * HD + d) * S_LEN + s] = f2bf(v);
        }
      }
}

// ---------------------------------------------------------------------------
// Flash attention (causal, GQA): one block = (b, h, 128-query tile),
// 4 waves * 32 rows. 64-key K/V tiles DMA'd by TDM (wave 0) into
// double-buffered LDS (72-elem stride: 32-DW rows + 4-DW pad); online softmax
// in exp2 domain; P converted C-layout -> A-layout via per-wave LDS buffer.
// ---------------------------------------------------------------------------
__global__ __launch_bounds__(128)
void flash_attn_wmma(const bf16_t* __restrict__ Q, const bf16_t* __restrict__ Kb,
                     const bf16_t* __restrict__ Vt, bf16_t* __restrict__ O) {
  __shared__ alignas(16) bf16_t lK[2][64 * 72];       // [key][d], padded
  __shared__ alignas(16) bf16_t lV[2][64 * 72];       // [d][key], padded
  __shared__ alignas(16) bf16_t lP[4][2 * 16 * 72];   // per-wave P buffers

  const int qt = blockIdx.x;                 // 0..15 query tiles
  const int bh = blockIdx.y;                 // 0..63 (b*32 + h)
  const int b = bh >> 5, h = bh & 31, hk = h >> 2;
  const int tid = threadIdx.x;
  const int lane = tid & 31, wid = tid >> 5;
  const int half = lane >> 4, l16 = lane & 15;
  const int qw0 = qt * 128 + wid * 32;       // wave's query base row

  const bf16_t* Qbase = Q  + ((size_t)b * HQ  + h ) * S_LEN * HD;
  const bf16_t* Kbase = Kb + ((size_t)b * HKV + hk) * S_LEN * HD;
  const bf16_t* Vbase = Vt + ((size_t)b * HKV + hk) * HD * S_LEN;
  const unsigned ldsK[2] = {(unsigned)(uintptr_t)&lK[0][0],
                            (unsigned)(uintptr_t)&lK[1][0]};
  const unsigned ldsV[2] = {(unsigned)(uintptr_t)&lV[0][0],
                            (unsigned)(uintptr_t)&lV[1][0]};

  const int ntiles = qt * 2 + 2;             // causal bound for this q-tile

  if (wid == 0) {                            // prime buffer 0
    tdm_load_bf16_2d(ldsK[0], Kbase, HD, 64, HD, 4, 3);     // 32-DW rows
    tdm_load_bf16_2d(ldsV[0], Vbase, 64, HD, S_LEN, 4, 3);
  }

  // Q fragments held in registers for whole kernel: [qb][D-chunk]
  v16bf qf[2][2];
#pragma unroll
  for (int qb = 0; qb < 2; ++qb)
#pragma unroll
    for (int kk = 0; kk < 2; ++kk) {
      const bf16_t* r = Qbase + (size_t)(qw0 + qb * 16 + l16) * HD
                        + kk * 32 + half * 8;
      qf[qb][kk] = mk16(*(const v8bf*)r, *(const v8bf*)(r + 16));
    }

  v8f oacc[2][4];
#pragma unroll
  for (int qb = 0; qb < 2; ++qb)
#pragma unroll
    for (int db = 0; db < 4; ++db)
#pragma unroll
      for (int e = 0; e < 8; ++e) oacc[qb][db][e] = 0.0f;

  float mrun[2][8], lrun[2][8];
#pragma unroll
  for (int qb = 0; qb < 2; ++qb)
#pragma unroll
    for (int e = 0; e < 8; ++e) { mrun[qb][e] = -1.0e30f; lrun[qb][e] = 0.0f; }

  const float SCL = 0.125f * 1.44269504f;    // 1/sqrt(64) * log2(e)

  for (int t = 0; t < ntiles; ++t) {
    const int k0 = t * 64;
    const int cur = t & 1;
    if (wid == 0) {
      if (t + 1 < ntiles) {    // prefetch next K/V tile into free buffer
        const int kn = (t + 1) * 64;
        tdm_load_bf16_2d(ldsK[cur ^ 1], Kbase + (size_t)kn * HD, HD, 64, HD, 4, 3);
        tdm_load_bf16_2d(ldsV[cur ^ 1], Vbase + kn, 64, HD, S_LEN, 4, 3);
        __builtin_amdgcn_s_wait_tensorcnt(2);
      } else {
        __builtin_amdgcn_s_wait_tensorcnt(0);
      }
    }
    __syncthreads();           // K/V tile `cur` visible to all waves

    if (k0 <= qw0 + 31) {      // wave-uniform: EXEC all-ones inside (WMMA req.)
#pragma unroll
      for (int qb = 0; qb < 2; ++qb) {
        // scores = Q(16x64) * K^T(64x64) over D, split into 4 key blocks
        v8f sc[4];
#pragma unroll
        for (int kb = 0; kb < 4; ++kb) {
#pragma unroll
          for (int e = 0; e < 8; ++e) sc[kb][e] = 0.0f;
#pragma unroll
          for (int kk = 0; kk < 2; ++kk) {
            const bf16_t* r = &lK[cur][(kb * 16 + l16) * 72 + kk * 32 + half * 8];
            v16bf bk = mk16(*(const v8bf*)r, *(const v8bf*)(r + 16));
            sc[kb] = wmma_bf16(qf[qb][kk], bk, sc[kb]);
          }
        }
        // scale, causal mask, row max (exp2 domain)
        float mx[8];
#pragma unroll
        for (int e = 0; e < 8; ++e) mx[e] = -1.0e30f;
#pragma unroll
        for (int kb = 0; kb < 4; ++kb)
#pragma unroll
          for (int e = 0; e < 8; ++e) {
            float v = sc[kb][e] * SCL;
            const int qrow = qw0 + qb * 16 + half * 8 + e;
            const int kcol = k0 + kb * 16 + l16;
            if (kcol > qrow) v = -1.0e30f;
            sc[kb][e] = v;
            mx[e] = fmaxf(mx[e], v);
          }
#pragma unroll
        for (int m = 1; m < 16; m <<= 1)
#pragma unroll
          for (int e = 0; e < 8; ++e)
            mx[e] = fmaxf(mx[e], __shfl_xor(mx[e], m, 32));

        float mnew[8], corr[8], ls[8];
#pragma unroll
        for (int e = 0; e < 8; ++e) {
          mnew[e] = fmaxf(mrun[qb][e], mx[e]);
          corr[e] = exp2f(mrun[qb][e] - mnew[e]);
          mrun[qb][e] = mnew[e];
          ls[e] = 0.0f;
        }
        // P = exp2(s - m); stash in LDS (C-layout -> A-layout round trip)
        bf16_t* Pb = &lP[wid][qb * 16 * 72];
#pragma unroll
        for (int kb = 0; kb < 4; ++kb)
#pragma unroll
          for (int e = 0; e < 8; ++e) {
            float pv = exp2f(sc[kb][e] - mnew[e]);
            ls[e] += pv;
            Pb[(half * 8 + e) * 72 + kb * 16 + l16] = f2bf(pv);
          }
#pragma unroll
        for (int m = 1; m < 16; m <<= 1)
#pragma unroll
          for (int e = 0; e < 8; ++e) ls[e] += __shfl_xor(ls[e], m, 32);
#pragma unroll
        for (int e = 0; e < 8; ++e)
          lrun[qb][e] = lrun[qb][e] * corr[e] + ls[e];
#pragma unroll
        for (int db = 0; db < 4; ++db)
#pragma unroll
          for (int e = 0; e < 8; ++e) oacc[qb][db][e] *= corr[e];

        // O += P(16x64 keys) * V(64 keys x 64 d)
#pragma unroll
        for (int kk = 0; kk < 2; ++kk) {
          const bf16_t* pr = Pb + l16 * 72 + kk * 32 + half * 8;
          v16bf pf = mk16(*(const v8bf*)pr, *(const v8bf*)(pr + 16));
#pragma unroll
          for (int db = 0; db < 4; ++db) {
            const bf16_t* r = &lV[cur][(db * 16 + l16) * 72 + kk * 32 + half * 8];
            v16bf bv = mk16(*(const v8bf*)r, *(const v8bf*)(r + 16));
            oacc[qb][db] = wmma_bf16(pf, bv, oacc[qb][db]);
          }
        }
      }
    }
    __syncthreads();           // buffer free before TDM overwrites it
  }

  // normalize + write attn output as [B, S, H*D] bf16 (row-major GEMM input)
#pragma unroll
  for (int qb = 0; qb < 2; ++qb)
#pragma unroll
    for (int db = 0; db < 4; ++db)
#pragma unroll
      for (int e = 0; e < 8; ++e) {
        const int qrow = qw0 + qb * 16 + half * 8 + e;
        const int d = db * 16 + l16;
        const float v = oacc[qb][db][e] / lrun[qb][e];
        O[((size_t)b * S_LEN + qrow) * EMBED + h * HD + d] = f2bf(v);
      }
}

// ---------------------------------------------------------------------------
// Host-side launcher
// ---------------------------------------------------------------------------
extern "C" void kernel_launch(void* const* d_in, const int* in_sizes, int n_in,
                              void* d_out, int out_size, void* d_ws, size_t ws_size,
                              hipStream_t stream) {
  (void)in_sizes; (void)n_in; (void)out_size; (void)ws_size;
  const float* x  = (const float*)d_in[0];
  const float* Wq = (const float*)d_in[1];
  const float* Wk = (const float*)d_in[2];
  const float* Wv = (const float*)d_in[3];
  const float* Wo = (const float*)d_in[4];

  bf16_t* p = (bf16_t*)d_ws;
  bf16_t* xb  = p;  p += (size_t)M_TOK * EMBED;   // x bf16
  bf16_t* wqT = p;  p += (size_t)EMBED * EMBED;   // Wq^T
  bf16_t* wkT = p;  p += (size_t)KVD   * EMBED;   // Wk^T
  bf16_t* wvT = p;  p += (size_t)KVD   * EMBED;   // Wv^T
  bf16_t* woT = p;  p += (size_t)EMBED * EMBED;   // Wo^T
  bf16_t* Qb  = p;  p += (size_t)M_TOK * EMBED;   // Q [B,H,S,D]
  bf16_t* Kbf = p;  p += (size_t)M_TOK * KVD;     // K [B,Hkv,S,D]
  bf16_t* Vtb = p;  p += (size_t)M_TOK * KVD;     // V^T [B,Hkv,D,S]
  bf16_t* Ab  = p;  p += (size_t)M_TOK * EMBED;   // attn out [B,S,E]

  {
    int n = M_TOK * EMBED;
    cvt_f32_bf16<<<n / (256 * 8), 256, 0, stream>>>(x, xb, n);
  }
  dim3 tb(32, 8);
  wtrans_bf16<<<dim3(EMBED / 32, EMBED / 32), tb, 0, stream>>>(Wq, wqT, EMBED, EMBED);
  wtrans_bf16<<<dim3(KVD   / 32, EMBED / 32), tb, 0, stream>>>(Wk, wkT, EMBED, KVD);
  wtrans_bf16<<<dim3(KVD   / 32, EMBED / 32), tb, 0, stream>>>(Wv, wvT, EMBED, KVD);
  wtrans_bf16<<<dim3(EMBED / 32, EMBED / 32), tb, 0, stream>>>(Wo, woT, EMBED, EMBED);

  gemm_bf16_wmma<0><<<dim3(EMBED / 128, M_TOK / 128), 256, 0, stream>>>(xb, wqT, (void*)Qb,  EMBED);
  gemm_bf16_wmma<1><<<dim3(KVD   / 128, M_TOK / 128), 256, 0, stream>>>(xb, wkT, (void*)Kbf, KVD);
  gemm_bf16_wmma<2><<<dim3(KVD   / 128, M_TOK / 128), 256, 0, stream>>>(xb, wvT, (void*)Vtb, KVD);

  flash_attn_wmma<<<dim3(S_LEN / 128, BATCH * HQ), 128, 0, stream>>>(Qb, Kbf, Vtb, Ab);

  gemm_bf16_wmma<3><<<dim3(EMBED / 128, M_TOK / 128), 256, 0, stream>>>(Ab, woT, d_out, EMBED);
}
